// ESN_17514876633614
// MI455X (gfx1250) — matrix operand order repaired and checked by reading, not compiled, not run on hardware
//
#include <hip/hip_runtime.h>
#include <hip/hip_bf16.h>

// ============================ problem constants ============================
#define ESN_N     2048
#define ESN_T     16384
#define ESN_WASH  200
#define ESN_OUT   (ESN_T - ESN_WASH)   // 16184

// ============================ launch geometry ==============================
#define NWG       64          // persistent workgroups (guaranteed co-resident)
#define NTHR      256         // 8 wave32 per WG
#define ROWS_WG   32          // rows of W per WG  (2 row-groups of 16)
#define KCHUNKS   64          // 2048 / 32 k-chunks per row

// ============================ LDS layout (dynamic) =========================
#define A_BYTES   (2 * KCHUNKS * 32 * 32)      // 131072 B
#define XBUF_OFF  A_BYTES                      // 2048 bf16 broadcast of x_t
#define PART_OFF  (A_BYTES + 4096)             // 2 rg * 4 ksec * 16 rows f32
#define LDS_BYTES (PART_OFF + 8 * 16 * 4)      // 135680 B (< 160KB: 2 WGs/WGP ok)

typedef __attribute__((ext_vector_type(16))) __bf16 v16bf;
typedef __attribute__((ext_vector_type(8)))  float  v8f;

// =========================== init kernel ===================================
__global__ void __launch_bounds__(NTHR) esn_init_kernel(float* __restrict__ out,
                                                        int out_n,
                                                        __bf16* __restrict__ xg,
                                                        int* __restrict__ bar) {
    const int idx    = blockIdx.x * blockDim.x + threadIdx.x;
    const int stride = gridDim.x * blockDim.x;
    for (int i = idx; i < out_n; i += stride)     out[i] = 0.0f;
    for (int i = idx; i < 2 * ESN_N; i += stride) xg[i] = (__bf16)0.0f;
    if (idx == 0) *bar = 0;
}

// ======================= device-wide ticket barrier ========================
__device__ __forceinline__ void grid_barrier(int* bar, int target) {
    __syncthreads();
    if (threadIdx.x == 0) {
        __threadfence();
        __hip_atomic_fetch_add(bar, 1, __ATOMIC_ACQ_REL, __HIP_MEMORY_SCOPE_AGENT);
        while (__hip_atomic_load(bar, __ATOMIC_ACQUIRE, __HIP_MEMORY_SCOPE_AGENT) < target)
            __builtin_amdgcn_s_sleep(1);
        __threadfence();
    }
    __syncthreads();
}

// ============================ main persistent kernel =======================
__global__ void __launch_bounds__(NTHR) esn_main_kernel(
        const float* __restrict__ u,
        const float* __restrict__ w_in,
        const float* __restrict__ w_res,
        const float* __restrict__ w_out,
        const float* __restrict__ w_mask,
        float* __restrict__ out,
        __bf16* __restrict__ xg,     // double-buffered state: [2][ESN_N] bf16
        int* __restrict__ bar) {
    extern __shared__ char smem[];

    const int tid  = threadIdx.x;
    const int lane = tid & 31;
    const int wv   = tid >> 5;      // wave id 0..7
    const int rg   = wv >> 2;       // row-group 0..1 (16 rows each)
    const int ks   = wv & 3;        // k-section 0..3 (16 chunks each)

    // ---- one-time: stage 32x2048 bf16 slice of W into LDS, pre-swizzled ---
    // into a contiguous-k WMMA A-operand layout (matches ldB): lane L holds row M=L&15,
// element e maps to K = e + 16*(L>>4).
    for (int idx = tid; idx < 2 * KCHUNKS * 32 * 16; idx += NTHR) {
        const int e   = idx & 15;
        const int ln  = (idx >> 4) & 31;
        const int c   = (idx >> 9) & 63;
        const int rgi = idx >> 15;
        const int K   = e + ((ln >> 4) << 4);
        const int row = blockIdx.x * ROWS_WG + rgi * 16 + (ln & 15);
        const int col = c * 32 + K;
        ((__bf16*)smem)[idx] = (__bf16)w_res[row * ESN_N + col];
    }

    float* part = (float*)(smem + PART_OFF);

    // Fragment fetch helpers (32B contiguous LDS reads each).
    auto ldA = [&](int c) -> v16bf {
        return *(const v16bf*)(smem + (((rg * KCHUNKS + c) * 32 + lane) << 5));
    };
    auto ldB = [&](int c) -> v16bf {
        return *(const v16bf*)(smem + XBUF_OFF + (c << 6) + ((lane & 16) << 1));
    };

    const int cbase = ks * 16;
    const int clast = cbase + 15;

    for (int t = 0; t < ESN_T; ++t) {
        // ---- broadcast x_t from global (L2-resident 4KB) into LDS --------
        const int4* xsrc = (const int4*)(xg + ((t & 1) ? ESN_N : 0));
        ((int4*)(smem + XBUF_OFF))[tid] = xsrc[tid];   // 256 * 16B = 4096B
        __syncthreads();

        // ---- GEMV slab via bf16 WMMA, f32 accumulate ---------------------
        // Software-pipelined: operands for chunk i+2/i+3 are fetched while
        // the WMMAs for chunks i/i+1 execute (hides LDS latency behind XDL).
        // Two accumulators break the serial D->C dependence chain.
        v16bf a0 = ldA(cbase + 0), b0 = ldB(cbase + 0);
        v16bf a1 = ldA(cbase + 1), b1 = ldB(cbase + 1);
        v8f acc0 = {}, acc1 = {};
        #pragma unroll
        for (int i = 0; i < 16; i += 2) {
            const int p0 = (cbase + i + 2 <= clast) ? (cbase + i + 2) : clast;
            const int p1 = (cbase + i + 3 <= clast) ? (cbase + i + 3) : clast;
            v16bf a2 = ldA(p0), b2 = ldB(p0);
            v16bf a3 = ldA(p1), b3 = ldB(p1);
            acc0 = __builtin_amdgcn_wmma_f32_16x16x32_bf16(
                       false, a0, false, b0, (short)0, acc0, false, false);
            acc1 = __builtin_amdgcn_wmma_f32_16x16x32_bf16(
                       false, a1, false, b1, (short)0, acc1, false, false);
            a0 = a2; b0 = b2; a1 = a3; b1 = b3;
        }
        const v8f acc = acc0 + acc1;

        // ---- extract row sums (all 16 D-columns identical) ---------------
        // lane 0 holds rows 0..7 in acc[0..7]; lane 16 holds rows 8..15.
        if ((lane & 15) == 0) {
            const int rb = lane >> 1;                    // 0 or 8
            #pragma unroll
            for (int i = 0; i < 8; ++i)
                part[(rg * 4 + ks) * 16 + rb + i] = acc[i];
        }
        __syncthreads();

        // ---- finalize 32 rows: cross-wave reduce, tanh, state + readout ---
        if (tid < 32) {
            const int r   = tid;
            const int rg2 = r >> 4, rr = r & 15;
            float s = part[(rg2 * 4 + 0) * 16 + rr]
                    + part[(rg2 * 4 + 1) * 16 + rr]
                    + part[(rg2 * 4 + 2) * 16 + rr]
                    + part[(rg2 * 4 + 3) * 16 + rr];
            const int grow = blockIdx.x * ROWS_WG + r;
            const float ut = u[t];
            const float xn = tanhf(fmaf(w_in[grow], ut, s));
            xg[((t & 1) ? 0 : ESN_N) + grow] = (__bf16)xn;   // write next buffer
            float yv = xn * w_mask[grow] * w_out[grow];
            #pragma unroll
            for (int off = 16; off > 0; off >>= 1)
                yv += __shfl_down(yv, off, 32);
            if (r == 0 && t >= ESN_WASH)
                atomicAdd(out + (t - ESN_WASH), yv);
            __threadfence();   // publish x/y stores before barrier arrival
        }

        // ---- device-wide step barrier ------------------------------------
        grid_barrier(bar, (t + 1) * NWG);
    }
}

// ================================ host side ================================
extern "C" void kernel_launch(void* const* d_in, const int* in_sizes, int n_in,
                              void* d_out, int out_size, void* d_ws, size_t ws_size,
                              hipStream_t stream) {
    const float* u      = (const float*)d_in[0];
    const float* w_in   = (const float*)d_in[1];
    const float* w_res  = (const float*)d_in[2];
    const float* w_out  = (const float*)d_in[3];
    const float* w_mask = (const float*)d_in[4];
    float* out          = (float*)d_out;

    __bf16* xg  = (__bf16*)d_ws;                      // 2*2048 bf16 = 8KB
    int*    bar = (int*)((char*)d_ws + 8192);         // barrier ticket

    (void)hipFuncSetAttribute(reinterpret_cast<const void*>(esn_main_kernel),
                              hipFuncAttributeMaxDynamicSharedMemorySize,
                              LDS_BYTES);

    esn_init_kernel<<<dim3(NWG), dim3(NTHR), 0, stream>>>(out, ESN_OUT, xg, bar);
    esn_main_kernel<<<dim3(NWG), dim3(NTHR), LDS_BYTES, stream>>>(
        u, w_in, w_res, w_out, w_mask, out, xg, bar);
}
